// BertAoA_Decoder_Core_10213432230228
// MI455X (gfx1250) — compile-verified, matrix operand order checked by reading, not run on hardware
//
#include <hip/hip_runtime.h>
#include <stdint.h>

// ---------------------------------------------------------------------------
// Problem constants
// ---------------------------------------------------------------------------
#define BB   64
#define SS   128
#define CC   196
#define DD   1024
#define HH   16
#define LL   6
#define FF   4096
#define DK   64
#define M1   (BB * SS)      // 8192
#define MC   (BB * CC)      // 12544

typedef _Float16 h16_t;
typedef __attribute__((ext_vector_type(16))) _Float16 v16h;
typedef __attribute__((ext_vector_type(8)))  _Float16 v8h;
typedef __attribute__((ext_vector_type(4)))  _Float16 v4h;
typedef __attribute__((ext_vector_type(8)))  float    v8f;
typedef int v4i __attribute__((vector_size(16)));

// ---------------------------------------------------------------------------
// CDNA5 async global->LDS copy (ASYNCcnt path), with safe fallback
// ---------------------------------------------------------------------------
#if defined(__HIP_DEVICE_COMPILE__) && __has_builtin(__builtin_amdgcn_global_load_async_to_lds_b128)
#define HAVE_ASYNC 1
#else
#define HAVE_ASYNC 0
#endif

__device__ __forceinline__ void copy16_to_lds(const h16_t* __restrict__ g,
                                              h16_t* __restrict__ l) {
#if HAVE_ASYNC
    __builtin_amdgcn_global_load_async_to_lds_b128(
        (__attribute__((address_space(1))) v4i*)(g),
        (__attribute__((address_space(3))) v4i*)(l),
        0 /*offset*/, 0 /*cpol*/);
#else
    *(v8h*)l = *(const v8h*)g;
#endif
}

__device__ __forceinline__ void async_wait0() {
#if HAVE_ASYNC
#if __has_builtin(__builtin_amdgcn_s_wait_asynccnt)
    __builtin_amdgcn_s_wait_asynccnt(0);
#else
    asm volatile("s_wait_asynccnt 0x0" ::: "memory");
#endif
#endif
}

// ---------------------------------------------------------------------------
// x = xt (float4 copy)
// ---------------------------------------------------------------------------
__global__ void copy_f4_kernel(const float* __restrict__ src, float* __restrict__ dst) {
    size_t i = (size_t)blockIdx.x * blockDim.x + threadIdx.x;
    ((float4*)dst)[i] = ((const float4*)src)[i];
}

// ---------------------------------------------------------------------------
// f32 -> f16 elementwise (vectorized x4)
// ---------------------------------------------------------------------------
__global__ void cvtf16_kernel(const float* __restrict__ in, h16_t* __restrict__ out) {
    size_t i = (size_t)blockIdx.x * blockDim.x + threadIdx.x;
    float4 f = ((const float4*)in)[i];
    v4h o = { (h16_t)f.x, (h16_t)f.y, (h16_t)f.z, (h16_t)f.w };
    *(v4h*)&out[i * 4] = o;
}

// ---------------------------------------------------------------------------
// Weight convert + transpose: W[K,N] f32  ->  Wt[N,K] f16  (32x32 LDS tiles)
// ---------------------------------------------------------------------------
__global__ void wconv_kernel(const float* __restrict__ in, h16_t* __restrict__ out,
                             int K, int N) {
    __shared__ float tile[32][33];
    const int n0 = blockIdx.x * 32;
    const int k0 = blockIdx.y * 32;
    const int tx = threadIdx.x;   // 0..31
    const int ty = threadIdx.y;   // 0..7
#pragma unroll
    for (int i = 0; i < 32; i += 8)
        tile[ty + i][tx] = in[(size_t)(k0 + ty + i) * N + (n0 + tx)];
    __syncthreads();
#pragma unroll
    for (int i = 0; i < 32; i += 8)
        out[(size_t)(n0 + ty + i) * K + (k0 + tx)] = (h16_t)tile[tx][ty + i];
}

// ---------------------------------------------------------------------------
// LayerNorm over D=1024 per row -> f16 output. 256 threads/block.
// ---------------------------------------------------------------------------
__global__ void ln_kernel(const float* __restrict__ x,
                          const float* __restrict__ g,
                          const float* __restrict__ b,
                          h16_t* __restrict__ out) {
    __shared__ float red1[256];
    __shared__ float red2[256];
    const int tid = threadIdx.x;
    const size_t row = blockIdx.x;
    const float* xr = x + row * (size_t)DD;
    float v[4];
    float s = 0.f, s2 = 0.f;
#pragma unroll
    for (int i = 0; i < 4; ++i) {
        v[i] = xr[tid + 256 * i];
        s += v[i];
        s2 += v[i] * v[i];
    }
    red1[tid] = s; red2[tid] = s2;
    __syncthreads();
    for (int off = 128; off > 0; off >>= 1) {
        if (tid < off) { red1[tid] += red1[tid + off]; red2[tid] += red2[tid + off]; }
        __syncthreads();
    }
    const float mu  = red1[0] * (1.0f / DD);
    const float var = red2[0] * (1.0f / DD) - mu * mu;
    const float inv = rsqrtf(var + 1e-6f);
    h16_t* orow = out + row * (size_t)DD;
#pragma unroll
    for (int i = 0; i < 4; ++i) {
        const int col = tid + 256 * i;
        orow[col] = (h16_t)((v[i] - mu) * inv * g[col] + b[col]);
    }
}

// ---------------------------------------------------------------------------
// WMMA GEMM: out = act( A[M,K](f16) @ Wt[N,K](f16)^T + bias (+ Res) )
// 128x128x32 tiles, 8 wave32, double-buffered LDS with async global->LDS.
// MODE 0: f16 out.  MODE 1: tanh-GELU, f16 out.  MODE 2: +Res(f32), f32 out.
// ---------------------------------------------------------------------------
__device__ __forceinline__ float gelu_tanh(float x) {
    const float c0 = 0.7978845608028654f;
    const float c1 = 0.044715f;
    return 0.5f * x * (1.0f + tanhf(c0 * (x + c1 * x * x * x)));
}

template <int MODE>
__global__ __launch_bounds__(256)
void gemm_wmma_kernel(const h16_t* __restrict__ A,
                      const h16_t* __restrict__ Wt,
                      const float* __restrict__ bias,
                      const float* __restrict__ Res,
                      void* __restrict__ OutV,
                      int M, int N, int K) {
    // Row stride 40 halves keeps every 8-half chunk 16B-aligned.
    __shared__ h16_t sA[2][128 * 40];
    __shared__ h16_t sB[2][128 * 40];

    const int tid  = threadIdx.x;
    const int wave = tid >> 5;
    const int lane = tid & 31;
    const int tileM = blockIdx.y * 128;
    const int tileN = blockIdx.x * 128;

    // Each thread stages two 16B chunks per tile (512 chunks = 128 rows x 4).
    const int c0 = tid, c1 = tid + 256;
    const int r0 = c0 >> 2, o0 = (c0 & 3) * 8;
    const int r1 = c1 >> 2, o1 = (c1 & 3) * 8;
    const h16_t* gA0 = A  + (size_t)(tileM + r0) * K + o0;
    const h16_t* gA1 = A  + (size_t)(tileM + r1) * K + o1;
    const h16_t* gB0 = Wt + (size_t)(tileN + r0) * K + o0;
    const h16_t* gB1 = Wt + (size_t)(tileN + r1) * K + o1;
    const int l0 = r0 * 40 + o0;
    const int l1 = r1 * 40 + o1;

    auto stage = [&](int buf, int k0) {
        copy16_to_lds(gA0 + k0, &sA[buf][l0]);
        copy16_to_lds(gA1 + k0, &sA[buf][l1]);
        copy16_to_lds(gB0 + k0, &sB[buf][l0]);
        copy16_to_lds(gB1 + k0, &sB[buf][l1]);
    };

    v8f acc[8];
#pragma unroll
    for (int t = 0; t < 8; ++t) acc[t] = (v8f)0.0f;

    stage(0, 0);
    async_wait0();
    __syncthreads();

    const int fm = lane & 15;          // fragment row/col within 16
    const int kb = (lane >> 4) * 8;    // K half-group base
    int buf = 0;

    for (int k0 = 0; k0 < K; k0 += 32) {
        if (k0 + 32 < K) stage(buf ^ 1, k0 + 32);

        const v8h alo = *(const v8h*)&sA[buf][(wave * 16 + fm) * 40 + kb];
        const v8h ahi = *(const v8h*)&sA[buf][(wave * 16 + fm) * 40 + 16 + kb];
        const v16h af = __builtin_shufflevector(alo, ahi,
            0, 1, 2, 3, 4, 5, 6, 7, 8, 9, 10, 11, 12, 13, 14, 15);

#pragma unroll
        for (int t = 0; t < 8; ++t) {
            const v8h blo = *(const v8h*)&sB[buf][(t * 16 + fm) * 40 + kb];
            const v8h bhi = *(const v8h*)&sB[buf][(t * 16 + fm) * 40 + 16 + kb];
            const v16h bf = __builtin_shufflevector(blo, bhi,
                0, 1, 2, 3, 4, 5, 6, 7, 8, 9, 10, 11, 12, 13, 14, 15);
            acc[t] = __builtin_amdgcn_wmma_f32_16x16x32_f16(
                false, af, false, bf, (short)0, acc[t], false, false);
        }

        async_wait0();
        __syncthreads();
        buf ^= 1;
    }

    // Epilogue. C layout: VGPR r -> M = r + 8*(lane>=16), N = lane%16.
    const int mm = (lane >> 4) * 8;
    const int nn = lane & 15;
    float* Of = (float*)OutV;
    h16_t* Oh = (h16_t*)OutV;
#pragma unroll
    for (int t = 0; t < 8; ++t) {
        const int col = tileN + t * 16 + nn;
        const float bv = bias[col];
#pragma unroll
        for (int r = 0; r < 8; ++r) {
            const int row = tileM + wave * 16 + r + mm;
            float v = acc[t][r] + bv;
            if (MODE == 1) v = gelu_tanh(v);
            if (MODE == 2) {
                v += Res[(size_t)row * N + col];
                Of[(size_t)row * N + col] = v;
            } else {
                Oh[(size_t)row * N + col] = (h16_t)v;
            }
        }
    }
}

// ---------------------------------------------------------------------------
// Attention: one block per (b, h, s). Q/K/V f16 [rows, D] head-interleaved.
// ---------------------------------------------------------------------------
__global__ __launch_bounds__(256)
void attn_kernel(const h16_t* __restrict__ Q,
                 const h16_t* __restrict__ Kb,
                 const h16_t* __restrict__ Vb,
                 h16_t* __restrict__ Out,
                 int Tk) {
    __shared__ float q[DK];
    __shared__ float sc[256];
    __shared__ float red[256];
    const int tid = threadIdx.x;
    const int s = blockIdx.x, h = blockIdx.y, b = blockIdx.z;
    const size_t qoff = ((size_t)(b * SS + s)) * DD + h * DK;

    if (tid < DK) q[tid] = (float)Q[qoff + tid];
    __syncthreads();

    float sval = -1e30f;
    if (tid < Tk) {
        const h16_t* kr = Kb + ((size_t)(b * Tk + tid)) * DD + h * DK;
        float d = 0.f;
#pragma unroll 8
        for (int i = 0; i < DK; ++i) d += q[i] * (float)kr[i];
        sval = d * 0.125f;   // 1/sqrt(64)
    }
    red[tid] = sval;
    __syncthreads();
    for (int off = 128; off > 0; off >>= 1) {
        if (tid < off) red[tid] = fmaxf(red[tid], red[tid + off]);
        __syncthreads();
    }
    const float mx = red[0];
    __syncthreads();

    const float e = (tid < Tk) ? __expf(sval - mx) : 0.f;
    sc[tid] = e;
    red[tid] = e;
    __syncthreads();
    for (int off = 128; off > 0; off >>= 1) {
        if (tid < off) red[tid] += red[tid + off];
        __syncthreads();
    }
    const float inv = 1.0f / red[0];
    __syncthreads();

    if (tid < DK) {
        const h16_t* vb = Vb + ((size_t)b * Tk) * DD + h * DK + tid;
        float acc = 0.f;
        for (int t = 0; t < Tk; ++t) acc += sc[t] * (float)vb[(size_t)t * DD];
        Out[qoff + tid] = (h16_t)(acc * inv);
    }
}

// ---------------------------------------------------------------------------
// Host orchestration
// ---------------------------------------------------------------------------
static inline void launch_gemm(int mode, const h16_t* A, const h16_t* Wt,
                               const float* bias, const float* Res, void* Out,
                               int M, int N, int K, hipStream_t stream) {
    dim3 grid(N / 128, M / 128);
    dim3 block(256);
    if (mode == 0)      hipLaunchKernelGGL((gemm_wmma_kernel<0>), grid, block, 0, stream, A, Wt, bias, Res, Out, M, N, K);
    else if (mode == 1) hipLaunchKernelGGL((gemm_wmma_kernel<1>), grid, block, 0, stream, A, Wt, bias, Res, Out, M, N, K);
    else                hipLaunchKernelGGL((gemm_wmma_kernel<2>), grid, block, 0, stream, A, Wt, bias, Res, Out, M, N, K);
}

extern "C" void kernel_launch(void* const* d_in, const int* in_sizes, int n_in,
                              void* d_out, int out_size, void* d_ws, size_t ws_size,
                              hipStream_t stream) {
    const float* xt    = (const float*)d_in[0];
    const float* p_att = (const float*)d_in[3];
    const float* Wq = (const float*)d_in[4];  const float* bq = (const float*)d_in[5];
    const float* Wk = (const float*)d_in[6];  const float* bk = (const float*)d_in[7];
    const float* Wv = (const float*)d_in[8];  const float* bv = (const float*)d_in[9];
    const float* Wo = (const float*)d_in[10]; const float* bo = (const float*)d_in[11];
    const float* W1 = (const float*)d_in[12]; const float* b1 = (const float*)d_in[13];
    const float* W2 = (const float*)d_in[14]; const float* b2 = (const float*)d_in[15];
    const float* ln1_g = (const float*)d_in[16]; const float* ln1_b = (const float*)d_in[17];
    const float* ln2_g = (const float*)d_in[18]; const float* ln2_b = (const float*)d_in[19];

    float* x = (float*)d_out;          // running residual stream, f32 [M1, D]

    // ---- workspace carve (all f16 except x) ----
    const size_t DxD = (size_t)DD * DD;          // 1M
    const size_t WT_HALVES = 4 * DxD + 2 * (size_t)DD * FF;  // 12M halves / layer
    h16_t* wt   = (h16_t*)d_ws;
    h16_t* p16  = wt  + WT_HALVES;
    h16_t* h16  = p16 + (size_t)MC * DD;
    h16_t* q16  = h16 + (size_t)M1 * DD;
    h16_t* k16  = q16 + (size_t)M1 * DD;
    h16_t* v16  = k16 + (size_t)MC * DD;
    h16_t* a16  = v16 + (size_t)MC * DD;
    h16_t* f16b = q16;   // FFN buffer [M1, F] aliases dead Q/K/V region

    h16_t* wq_t = wt;
    h16_t* wk_t = wt + DxD;
    h16_t* wv_t = wt + 2 * DxD;
    h16_t* wo_t = wt + 3 * DxD;
    h16_t* w1_t = wt + 4 * DxD;                       // [F, D]
    h16_t* w2_t = wt + 4 * DxD + (size_t)DD * FF;     // [D, F]

    // x = xt
    {
        const size_t n4 = (size_t)M1 * DD / 4;
        hipLaunchKernelGGL(copy_f4_kernel, dim3((unsigned)(n4 / 256)), dim3(256), 0, stream, xt, x);
    }
    // p16 = f16(p_att_feats)
    {
        const size_t n4 = (size_t)MC * DD / 4;
        hipLaunchKernelGGL(cvtf16_kernel, dim3((unsigned)(n4 / 256)), dim3(256), 0, stream, p_att, p16);
    }

    auto conv = [&](const float* w, h16_t* dst, int K, int N) {
        hipLaunchKernelGGL(wconv_kernel, dim3(N / 32, K / 32), dim3(32, 8), 0, stream, w, dst, K, N);
    };

    for (int i = 0; i < LL; ++i) {
        const size_t wOff = (size_t)i * DxD;
        const size_t bOff = (size_t)i * DD;

        // per-layer weight convert+transpose into the shared f16 buffer
        conv(Wq + wOff, wq_t, DD, DD);
        conv(Wk + wOff, wk_t, DD, DD);
        conv(Wv + wOff, wv_t, DD, DD);
        conv(Wo + wOff, wo_t, DD, DD);
        conv(W1 + (size_t)i * DD * FF, w1_t, DD, FF);
        conv(W2 + (size_t)i * FF * DD, w2_t, FF, DD);

        // h = LN1(x)
        hipLaunchKernelGGL(ln_kernel, dim3(M1), dim3(256), 0, stream,
                           x, ln1_g + bOff, ln1_b + bOff, h16);

        const h16_t* kvsrc = (i == 0) ? p16 : h16;
        const int Tk  = (i == 0) ? CC : SS;
        const int Mkv = (i == 0) ? MC : M1;

        launch_gemm(0, h16,   wq_t, bq + bOff, nullptr, q16, M1,  DD, DD, stream);
        launch_gemm(0, kvsrc, wk_t, bk + bOff, nullptr, k16, Mkv, DD, DD, stream);
        launch_gemm(0, kvsrc, wv_t, bv + bOff, nullptr, v16, Mkv, DD, DD, stream);

        hipLaunchKernelGGL(attn_kernel, dim3(SS, HH, BB), dim3(256), 0, stream,
                           q16, k16, v16, a16, Tk);

        // x = x + a @ Wo + bo
        launch_gemm(2, a16, wo_t, bo + bOff, x, x, M1, DD, DD, stream);

        // h = LN2(x)
        hipLaunchKernelGGL(ln_kernel, dim3(M1), dim3(256), 0, stream,
                           x, ln2_g + bOff, ln2_b + bOff, h16);

        // f = gelu(h @ W1 + b1)
        launch_gemm(1, h16, w1_t, b1 + (size_t)i * FF, nullptr, f16b, M1, FF, DD, stream);

        // x = x + f @ W2 + b2
        launch_gemm(2, f16b, w2_t, b2 + bOff, x, x, M1, DD, FF, stream);
    }
}
